// NeuralLanguageGenerator_67954972557595
// MI455X (gfx1250) — matrix operand order repaired and checked by reading, not compiled, not run on hardware
//
#include <hip/hip_runtime.h>

// ---------------------------------------------------------------------------
// Types for CDNA5 WMMA (wave32)
// ---------------------------------------------------------------------------
typedef __bf16 bf16;
typedef __attribute__((ext_vector_type(16))) bf16  v16bf;
typedef __attribute__((ext_vector_type(8)))  bf16  v8bf;
typedef __attribute__((ext_vector_type(8)))  float v8f;
typedef __attribute__((ext_vector_type(4)))  int   v4i;

#define PI2 6.28318530717958647692f

// ---------------------------------------------------------------------------
// CDNA5 async global->LDS path (ASYNCcnt), guarded so either toolchain builds.
// Probe round 2 confirmed: builtin exists, signature is
//   (int4 addrspace(1)*, int4 addrspace(3)*, imm offset, imm cpol)
// ---------------------------------------------------------------------------
#if defined(__has_builtin)
#if __has_builtin(__builtin_amdgcn_global_load_async_to_lds_b128)
#define HAVE_ASYNC_LDS 1
#endif
#endif
#ifndef HAVE_ASYNC_LDS
#define HAVE_ASYNC_LDS 0
#endif

#if HAVE_ASYNC_LDS
#if __has_builtin(__builtin_amdgcn_s_wait_asynccnt)
#define WAIT_ASYNCCNT(n) __builtin_amdgcn_s_wait_asynccnt(n)
#else
#define WAIT_ASYNCCNT(n) asm volatile("s_wait_asynccnt " #n ::: "memory")
#endif
#else
#define WAIT_ASYNCCNT(n)
#endif

// ---------------------------------------------------------------------------
// Model dims
// ---------------------------------------------------------------------------
static constexpr int E      = 256;
static constexpr int S      = 512;
static constexpr int B      = 4;
static constexpr int M_TOK  = B * S;          // 2048 tokens
static constexpr int VOCAB  = 50000;

// GEMM tile config
static constexpr int BM = 128;
static constexpr int BN = 128;
static constexpr int BK = 32;
static constexpr int LDSP = 40;               // bf16 elems per LDS row (32 + 8 pad = 80B pitch)

enum { ACT_NONE = 0, ACT_RELU = 1, ACT_GELU = 2, ACT_SIGN = 3 };

// ---------------------------------------------------------------------------
// Spectral filter -> real circulant matrix Mt[n][j] = r[(n-j) & 255]  (bf16)
// r[d] = (1/256) * sum_m cos(theta_m + 2*pi*m*d/256),
// theta_m = 1.5*atan(ln(k_m + 1e-6)), k_m = (4m<=512) ? 4m : 1024-4m
// ---------------------------------------------------------------------------
__global__ __launch_bounds__(256) void spectral_init_kernel(bf16* __restrict__ Mt) {
    __shared__ float r[256];
    const int d = threadIdx.x;
    float sum = 0.f;
    for (int m = 0; m < 256; ++m) {
        const int  km4 = 4 * m;
        const float k  = (km4 <= 512) ? (float)km4 : (float)(1024 - km4);
        const float th = 1.5f * atanf(logf(k + 1e-6f));
        const float ph = (float)((m * d) & 255) * (PI2 / 256.f);
        sum += cosf(th + ph);
    }
    r[d] = sum * (1.f / 256.f);
    __syncthreads();
    for (int n = 0; n < 256; ++n)
        Mt[n * 256 + d] = (bf16)r[(n - d) & 255];
}

// ---------------------------------------------------------------------------
// Embedding gather -> |v| (bf16) and sign(v) (f32), per token
// ---------------------------------------------------------------------------
__global__ __launch_bounds__(256) void gather_abs_sign_kernel(
    const int* __restrict__ ids, const float* __restrict__ emb,
    bf16* __restrict__ absv, float* __restrict__ sgn) {
    const int m = blockIdx.x;       // token 0..2047
    const int j = threadIdx.x;      // feature 0..255
    const float v = emb[(size_t)ids[m] * E + j];
    absv[(size_t)m * E + j] = (bf16)fabsf(v);
    sgn [(size_t)m * E + j] = (v > 0.f) ? 1.f : ((v < 0.f) ? -1.f : 0.f);
}

// ---------------------------------------------------------------------------
// Weight prep: out[c][k] = sum_{q<fold} in[k + K*q][c], f32 -> bf16,
// LDS 32x32 tiled transpose so both global phases are coalesced.
// in: [fold*K, C] row-major f32 ; out: [C, K] row-major bf16
// ---------------------------------------------------------------------------
__global__ __launch_bounds__(256) void fold_transpose_bf16_kernel(
    const float* __restrict__ in, bf16* __restrict__ out,
    int K, int C, int fold) {
    __shared__ float tile[32][33];
    const int k0 = blockIdx.x * 32, c0 = blockIdx.y * 32;
    const int tx = threadIdx.x & 31, ty = threadIdx.x >> 5;   // 32 x 8
    for (int i = 0; i < 4; ++i) {
        const int k = k0 + ty + i * 8, c = c0 + tx;
        float acc = 0.f;
        if (k < K && c < C)
            for (int q = 0; q < fold; ++q)
                acc += in[(size_t)(k + K * q) * C + c];
        tile[ty + i * 8][tx] = acc;
    }
    __syncthreads();
    for (int i = 0; i < 4; ++i) {
        const int c = c0 + ty + i * 8, k = k0 + tx;
        if (c < C && k < K)
            out[(size_t)c * K + k] = (bf16)tile[tx][ty + i * 8];
    }
}

// ---------------------------------------------------------------------------
// One 16B staging chunk: async global->LDS when available, else VGPR bounce.
// ---------------------------------------------------------------------------
__device__ __forceinline__ void stage16(const bf16* __restrict__ g, bf16* l) {
#if HAVE_ASYNC_LDS
    __builtin_amdgcn_global_load_async_to_lds_b128(
        (__attribute__((address_space(1))) v4i*)(bf16*)g,
        (__attribute__((address_space(3))) v4i*)l, 0, 0);
#else
    *(uint4*)l = *(const uint4*)g;
#endif
}

// ---------------------------------------------------------------------------
// Tiled bf16 WMMA GEMM:  out[M,N] = act(A[M,K] @ Bt[N,K]^T + bias)
//   - 256 threads = 8 waves; wave (wm,wn) owns 32x64 = 2x4 WMMA 16x16 tiles
//   - K multiple of 32, M multiple of 128; N guarded
//   - Double-buffered LDS; async global->LDS pipeline when builtin exists:
//     tile t+1 streams (ASYNCcnt) while tile t feeds v_wmma_f32_16x16x32_bf16
//   - Fragment packing follows CDNA5 ISA 7.12.2 (wave32)
// ---------------------------------------------------------------------------
template <int ACT, bool OUT_BF16>
__global__ __launch_bounds__(256) void gemm_bf16_wmma_kernel(
    const bf16*  __restrict__ A,      // [M,K]
    const bf16*  __restrict__ Bt,     // [N,K] (B transposed)
    const float* __restrict__ bias,   // [N] or nullptr
    const float* __restrict__ signm,  // [M,N] for ACT_SIGN, else nullptr
    float* __restrict__ outF, bf16* __restrict__ outB,
    int M, int N, int K) {
    __shared__ __align__(16) bf16 As[2][BM * LDSP];
    __shared__ __align__(16) bf16 Bs[2][BN * LDSP];

    const int tid   = threadIdx.x;
    const int lane  = tid & 31;
    const int wave  = tid >> 5;
    const int wm    = wave >> 1;     // 0..3 -> 32-row strip
    const int wn    = wave & 1;      // 0..1 -> 64-col strip
    const int lh    = lane >> 4;     // lane half 0/1
    const int lm    = lane & 15;
    const int bm    = blockIdx.y * BM;
    const int bn    = blockIdx.x * BN;

    // staging coords: thread owns 16B chunks at rows {srow, srow+64}, col scol
    const int srow  = tid >> 2;
    const int scol  = (tid & 3) * 8;
    const bool bfull = (bn + BN <= N);   // uniform per block

    v8f acc[2][4];
    const v8f vzero = {0.f, 0.f, 0.f, 0.f, 0.f, 0.f, 0.f, 0.f};
    for (int i = 0; i < 2; ++i)
        for (int j = 0; j < 4; ++j) acc[i][j] = vzero;

    auto stageA = [&](int k0, int buf) {
#pragma unroll
        for (int c = 0; c < 2; ++c) {
            const int row = srow + c * 64;
            stage16(A + (size_t)(bm + row) * K + k0 + scol,
                    &As[buf][row * LDSP + scol]);
        }
    };
    auto stageB = [&](int k0, int buf) {
        if (bfull) {
#pragma unroll
            for (int c = 0; c < 2; ++c) {
                const int row = srow + c * 64;
                stage16(Bt + (size_t)(bn + row) * K + k0 + scol,
                        &Bs[buf][row * LDSP + scol]);
            }
        } else {
#pragma unroll
            for (int c = 0; c < 2; ++c) {
                const int row = srow + c * 64;
                uint4 v = {0u, 0u, 0u, 0u};
                if (bn + row < N)
                    v = *(const uint4*)(Bt + (size_t)(bn + row) * K + k0 + scol);
                *(uint4*)(&Bs[buf][row * LDSP + scol]) = v;
            }
        }
    };

    const int nt = K / BK;
    stageA(0, 0);
    stageB(0, 0);

    for (int t = 0; t < nt; ++t) {
        const int buf = t & 1;
        const bool more = (t + 1 < nt);
        if (more) {                      // prefetch next tile into other buffer
            stageA((t + 1) * BK, buf ^ 1);
            stageB((t + 1) * BK, buf ^ 1);
        }
#if HAVE_ASYNC_LDS
        // async loads retire in order: <=4 (or <=2) leaves only next tile in flight
        if (bfull) { if (more) { WAIT_ASYNCCNT(4); } else { WAIT_ASYNCCNT(0); } }
        else       { if (more) { WAIT_ASYNCCNT(2); } else { WAIT_ASYNCCNT(0); } }
#endif
        __syncthreads();                 // all waves' tile t resident in LDS

        // ---- A fragments: lane (lh,lm): e<8 -> K=lh*8+e ; e>=8 -> K=16+lh*8+(e-8)
        v16bf afrag[2];
        for (int i = 0; i < 2; ++i) {
            const int ml = wm * 32 + i * 16 + lm;
            const v8bf lo = *(const v8bf*)&As[buf][ml * LDSP + lh * 8];
            const v8bf hi = *(const v8bf*)&As[buf][ml * LDSP + 16 + lh * 8];
            for (int e = 0; e < 8; ++e) { afrag[i][e] = lo[e]; afrag[i][8 + e] = hi[e]; }
        }
        // ---- B fragments: lane (lh,lm): e<8 -> K=lh*8+e ; e>=8 -> K=16+lh*8+(e-8) at column N=lm
        v16bf bfrag[4];
        for (int j = 0; j < 4; ++j) {
            const int nl = wn * 64 + j * 16 + lm;
            const v8bf lo = *(const v8bf*)&Bs[buf][nl * LDSP + lh * 8];
            const v8bf hi = *(const v8bf*)&Bs[buf][nl * LDSP + 16 + lh * 8];
            for (int e = 0; e < 8; ++e) { bfrag[j][e] = lo[e]; bfrag[j][8 + e] = hi[e]; }
        }

        for (int i = 0; i < 2; ++i)
            for (int j = 0; j < 4; ++j)
                acc[i][j] = __builtin_amdgcn_wmma_f32_16x16x32_bf16(
                    false, afrag[i], false, bfrag[j], (short)0, acc[i][j], false, false);
        __syncthreads();                 // reads of buf done before it is refilled
    }

    // ---- epilogue: C/D layout: VGPR r, lane (lh,lm) -> row = lh*8 + r, col = lm
    for (int j = 0; j < 4; ++j) {
        const int col = bn + wn * 64 + j * 16 + lm;
        if (col >= N) continue;
        const float bval = bias ? bias[col] : 0.f;
        for (int i = 0; i < 2; ++i) {
            for (int r = 0; r < 8; ++r) {
                const int row = bm + wm * 32 + i * 16 + lh * 8 + r;
                float v = acc[i][j][r] + bval;
                if (ACT == ACT_RELU)      v = fmaxf(v, 0.f);
                else if (ACT == ACT_GELU) v = 0.5f * v * (1.f + erff(v * 0.70710678118654752f));
                else if (ACT == ACT_SIGN) v = v * signm[(size_t)row * N + col];
                if (OUT_BF16) outB[(size_t)row * N + col] = (bf16)v;
                else          outF[(size_t)row * N + col] = v;
            }
        }
    }
}

// ---------------------------------------------------------------------------
// Row LayerNorm over 512 features, f32 in -> bf16 out
// ---------------------------------------------------------------------------
__global__ __launch_bounds__(256) void layernorm512_kernel(
    const float* __restrict__ h, const float* __restrict__ g,
    const float* __restrict__ b, bf16* __restrict__ out) {
    __shared__ float r1[256], r2[256];
    const int row = blockIdx.x, t = threadIdx.x;
    const float x0 = h[(size_t)row * 512 + t];
    const float x1 = h[(size_t)row * 512 + 256 + t];
    r1[t] = x0 + x1;
    r2[t] = x0 * x0 + x1 * x1;
    __syncthreads();
    for (int s = 128; s > 0; s >>= 1) {
        if (t < s) { r1[t] += r1[t + s]; r2[t] += r2[t + s]; }
        __syncthreads();
    }
    const float mu  = r1[0] * (1.f / 512.f);
    const float var = r2[0] * (1.f / 512.f) - mu * mu;
    const float inv = rsqrtf(var + 1e-5f);
    out[(size_t)row * 512 + t]       = (bf16)((x0 - mu) * inv * g[t]       + b[t]);
    out[(size_t)row * 512 + 256 + t] = (bf16)((x1 - mu) * inv * g[t + 256] + b[t + 256]);
}

// ---------------------------------------------------------------------------
// Mean over sequence -> concept_vector[b][32]; then structure head [4][8]
// One block per batch element.
// ---------------------------------------------------------------------------
__global__ __launch_bounds__(256) void concept_structure_kernel(
    const float* __restrict__ c3,    // [2048, 32] f32
    const float* __restrict__ sw1, const float* __restrict__ sb1,
    const float* __restrict__ sw2, const float* __restrict__ sb2,
    float* __restrict__ cv_out,      // [4, 32]
    float* __restrict__ sp_out) {    // [4, 8]
    __shared__ float part[256];
    __shared__ float cv[32];
    __shared__ float t1[16];
    const int b = blockIdx.x, t = threadIdx.x;
    const int dim = t & 31, sc = t >> 5;       // 8 seq-chunks x 32 dims
    float sum = 0.f;
    for (int s = sc; s < S; s += 8)
        sum += c3[((size_t)b * S + s) * 32 + dim];
    part[t] = sum;
    __syncthreads();
    if (t < 32) {
        float tot = 0.f;
        for (int q = 0; q < 8; ++q) tot += part[t + q * 32];
        const float m = tot * (1.f / (float)S);
        cv[t] = m;
        cv_out[b * 32 + t] = m;
    }
    __syncthreads();
    if (t < 16) {
        float a = sb1[t];
        for (int i = 0; i < 32; ++i) a += cv[i] * sw1[i * 16 + t];
        t1[t] = fmaxf(a, 0.f);
    }
    __syncthreads();
    if (t < 8) {
        float a = sb2[t];
        for (int n = 0; n < 16; ++n) a += t1[n] * sw2[n * 8 + t];
        sp_out[b * 8 + t] = 1.f / (1.f + expf(-a));
    }
}

// ---------------------------------------------------------------------------
// Host-side orchestration
// ---------------------------------------------------------------------------
static inline dim3 gemm_grid(int Mn, int Nn) {
    return dim3((Nn + BN - 1) / BN, Mn / BM);
}

extern "C" void kernel_launch(void* const* d_in, const int* in_sizes, int n_in,
                              void* d_out, int out_size, void* d_ws, size_t ws_size,
                              hipStream_t stream) {
    (void)in_sizes; (void)n_in; (void)out_size; (void)ws_size;

    const int*   char_ids = (const int*)  d_in[0];
    const float* emb      = (const float*)d_in[1];
    const float* dw1      = (const float*)d_in[2];
    const float* db1      = (const float*)d_in[3];
    const float* ln_g     = (const float*)d_in[4];
    const float* ln_b     = (const float*)d_in[5];
    const float* dw2      = (const float*)d_in[6];
    const float* db2      = (const float*)d_in[7];
    const float* dw3      = (const float*)d_in[8];
    const float* db3      = (const float*)d_in[9];
    const float* cw1      = (const float*)d_in[10];
    const float* cb1      = (const float*)d_in[11];
    const float* cw2      = (const float*)d_in[12];
    const float* cb2      = (const float*)d_in[13];
    const float* cw3      = (const float*)d_in[14];
    const float* cb3      = (const float*)d_in[15];
    const float* sw1      = (const float*)d_in[16];
    const float* sb1      = (const float*)d_in[17];
    const float* sw2      = (const float*)d_in[18];
    const float* sb2      = (const float*)d_in[19];

    // ---- workspace layout (all offsets 256B-aligned) ----
    char* w = (char*)d_ws;
    bf16*  Mt    = (bf16*) (w + 0);           // 256x256 bf16        131,072
    bf16*  absv  = (bf16*) (w + 131072);      // 2048x256 bf16     1,048,576
    float* sgn   = (float*)(w + 1179648);     // 2048x256 f32      2,097,152
    bf16*  sspec = (bf16*) (w + 3276800);     // 2048x256 bf16     1,048,576
    bf16*  dw1ft = (bf16*) (w + 4325376);     // 512x256 bf16        262,144
    bf16*  cw1ft = (bf16*) (w + 4587520);     // 256x256 bf16        131,072
    bf16*  dw2t  = (bf16*) (w + 4718592);     // 256x512 bf16        262,144
    bf16*  cw2t  = (bf16*) (w + 4980736);     // 64x256 bf16          32,768
    bf16*  cw3t  = (bf16*) (w + 5013504);     // 32x64 bf16            4,096
    bf16*  dw3t  = (bf16*) (w + 5017600);     // 50000x256 bf16   25,600,000
    bf16*  c1    = (bf16*) (w + 30617600);    // 2048x256 bf16     1,048,576
    bf16*  c2    = (bf16*) (w + 31666176);    // 2048x64 bf16        262,144
    float* c3    = (float*)(w + 31928320);    // 2048x32 f32         262,144
    float* h1    = (float*)(w + 32190464);    // 2048x512 f32      4,194,304
    bf16*  h1n   = (bf16*) (w + 36384768);    // 2048x512 bf16     2,097,152
    bf16*  h2    = (bf16*) (w + 38481920);    // 2048x256 bf16     1,048,576

    float* logits = (float*)d_out;                         // [2048, 50000]
    float* cv_out = logits + (size_t)M_TOK * VOCAB;        // [4, 32]
    float* sp_out = cv_out + B * 32;                       // [4, 8]

    // ---- prep: circulant spectral matrix, gather, weight fold/transpose ----
    spectral_init_kernel<<<1, 256, 0, stream>>>(Mt);
    gather_abs_sign_kernel<<<M_TOK, 256, 0, stream>>>(char_ids, emb, absv, sgn);

    fold_transpose_bf16_kernel<<<dim3(8, 16),   256, 0, stream>>>(dw1, dw1ft, 256, 512,   4);
    fold_transpose_bf16_kernel<<<dim3(8, 8),    256, 0, stream>>>(cw1, cw1ft, 256, 256,   4);
    fold_transpose_bf16_kernel<<<dim3(16, 8),   256, 0, stream>>>(dw2, dw2t,  512, 256,   1);
    fold_transpose_bf16_kernel<<<dim3(8, 2),    256, 0, stream>>>(cw2, cw2t,  256, 64,    1);
    fold_transpose_bf16_kernel<<<dim3(2, 1),    256, 0, stream>>>(cw3, cw3t,  64,  32,    1);
    fold_transpose_bf16_kernel<<<dim3(8, 1563), 256, 0, stream>>>(dw3, dw3t,  256, VOCAB, 1);

    // ---- spectral filter as GEMM: s = (|v| @ Mt^T) * sign(v) ----
    gemm_bf16_wmma_kernel<ACT_SIGN, true><<<gemm_grid(M_TOK, 256), 256, 0, stream>>>(
        absv, Mt, nullptr, sgn, nullptr, sspec, M_TOK, 256, 256);

    // ---- concept recognizer ----
    gemm_bf16_wmma_kernel<ACT_RELU, true><<<gemm_grid(M_TOK, 256), 256, 0, stream>>>(
        sspec, cw1ft, cb1, nullptr, nullptr, c1, M_TOK, 256, 256);
    gemm_bf16_wmma_kernel<ACT_RELU, true><<<gemm_grid(M_TOK, 64), 256, 0, stream>>>(
        c1, cw2t, cb2, nullptr, nullptr, c2, M_TOK, 64, 256);
    gemm_bf16_wmma_kernel<ACT_NONE, false><<<gemm_grid(M_TOK, 32), 256, 0, stream>>>(
        c2, cw3t, cb3, nullptr, c3, nullptr, M_TOK, 32, 64);
    concept_structure_kernel<<<B, 256, 0, stream>>>(c3, sw1, sb1, sw2, sb2, cv_out, sp_out);

    // ---- language decoder ----
    gemm_bf16_wmma_kernel<ACT_GELU, false><<<gemm_grid(M_TOK, 512), 256, 0, stream>>>(
        sspec, dw1ft, db1, nullptr, h1, nullptr, M_TOK, 512, 256);
    layernorm512_kernel<<<M_TOK, 256, 0, stream>>>(h1, ln_g, ln_b, h1n);
    gemm_bf16_wmma_kernel<ACT_GELU, true><<<gemm_grid(M_TOK, 256), 256, 0, stream>>>(
        h1n, dw2t, db2, nullptr, nullptr, h2, M_TOK, 256, 512);
    gemm_bf16_wmma_kernel<ACT_NONE, false><<<gemm_grid(M_TOK, VOCAB), 256, 0, stream>>>(
        h2, dw3t, db3, nullptr, logits, nullptr, M_TOK, VOCAB, 256);
}
